// GSA_32469952757971
// MI455X (gfx1250) — compile-verified, matrix operand order checked by reading, not run on hardware
//
#include <hip/hip_runtime.h>

typedef __attribute__((ext_vector_type(16))) _Float16 v16h;
typedef __attribute__((ext_vector_type(8)))  _Float16 v8h;
typedef __attribute__((ext_vector_type(8)))  float    v8f;

#define NB    4
#define OUTC  64
#define H0    128
#define W0    128
#define HW0   (H0 * W0)

// ---------------------------------------------------------------------------
// Pack NCHW f32 -> [B*H*W][C] f16 via LDS transpose.
// Tile: 64 channels x 32 positions. Reads coalesced along P (consecutive
// lanes -> consecutive addresses), writes coalesced along C. LDS padded to
// avoid bank conflicts on the transposed read-back.
// grid.x = B * HW/32 (position tiles), grid.y = C/64 (channel tiles)
// ---------------------------------------------------------------------------
__global__ void pack_a_kernel(const float* __restrict__ x, _Float16* __restrict__ A,
                              int C, int HW) {
  __shared__ _Float16 lds[64][34];
  int ptiles = HW >> 5;
  int b  = blockIdx.x / ptiles;
  int p0 = (blockIdx.x - b * ptiles) << 5;
  int c0 = blockIdx.y << 6;

  // read phase: lane -> position (coalesced), warp row -> channel
  int pl = threadIdx.x & 31;
  int cr = threadIdx.x >> 5;              // 0..7
  const float* src = x + ((size_t)(b * C + c0 + cr)) * HW + p0 + pl;
#pragma unroll
  for (int i = 0; i < 8; ++i)
    lds[cr + i * 8][pl] = (_Float16)src[(size_t)i * 8 * HW];
  __syncthreads();

  // write phase: lane -> channel (coalesced 64 consecutive f16 = 128B)
  int cl = threadIdx.x & 63;
  int pr = threadIdx.x >> 6;              // 0..3
  _Float16* dst = A + ((size_t)(b * HW + p0 + pr)) * C + c0 + cl;
#pragma unroll
  for (int i = 0; i < 8; ++i)
    dst[(size_t)i * 4 * C] = lds[cl][pr + i * 4];
}

__global__ void pack_w_kernel(const float* __restrict__ w, _Float16* __restrict__ Wh, int total) {
  int idx = blockIdx.x * blockDim.x + threadIdx.x;
  if (idx >= total) return;
  Wh[idx] = (_Float16)w[idx];
}

// ---------------------------------------------------------------------------
// WMMA fragment load: 16-bit operand, row r = lane&15, K-chunks at
// kbase(={0,8} by lane-half) and kbase+16  (ISA 7.12.2 16-bit A/B layout)
// ---------------------------------------------------------------------------
__device__ __forceinline__ v16h load_frag16(const _Float16* p) {
  v8h lo = *(const v8h*)(p);
  v8h hi = *(const v8h*)(p + 16);
  return __builtin_shufflevector(lo, hi, 0, 1, 2, 3, 4, 5, 6, 7, 8, 9, 10, 11, 12, 13, 14, 15);
}

// ---------------------------------------------------------------------------
// 1x1 conv as GEMM:  Y[b, n, p] = sum_k A[b*HW+p, k] * W[n, c_off + k]
// One wave -> 16(M) x 64(N) strip, 4 f32 accumulators, K stepped by 32.
// Stores straight into NCHW f32 (8 consecutive floats per lane, 32B aligned).
// ---------------------------------------------------------------------------
__global__ void conv1x1_wmma_kernel(const _Float16* __restrict__ A,
                                    const _Float16* __restrict__ Wh,
                                    float* __restrict__ Y,
                                    int K, int c_off, int HW) {
  int lane = threadIdx.x & 31;
  int wave = threadIdx.x >> 5;
  int m0   = (blockIdx.x * 8 + wave) * 16;
  int r    = lane & 15;
  int kb   = (lane & 16) ? 8 : 0;

  const _Float16* arow  = A  + (size_t)(m0 + r) * K + kb;
  const _Float16* brow0 = Wh + (size_t)r * 960 + c_off + kb;
  const _Float16* brow1 = brow0 + 16 * 960;
  const _Float16* brow2 = brow0 + 32 * 960;
  const _Float16* brow3 = brow0 + 48 * 960;

  v8f acc0 = {}; v8f acc1 = {}; v8f acc2 = {}; v8f acc3 = {};
  for (int k0 = 0; k0 < K; k0 += 32) {
    v16h a  = load_frag16(arow + k0);
    v16h b0 = load_frag16(brow0 + k0);
    v16h b1 = load_frag16(brow1 + k0);
    v16h b2 = load_frag16(brow2 + k0);
    v16h b3 = load_frag16(brow3 + k0);
    acc0 = __builtin_amdgcn_wmma_f32_16x16x32_f16(false, a, false, b0, (short)0, acc0, false, false);
    acc1 = __builtin_amdgcn_wmma_f32_16x16x32_f16(false, a, false, b1, (short)0, acc1, false, false);
    acc2 = __builtin_amdgcn_wmma_f32_16x16x32_f16(false, a, false, b2, (short)0, acc2, false, false);
    acc3 = __builtin_amdgcn_wmma_f32_16x16x32_f16(false, a, false, b3, (short)0, acc3, false, false);
  }

  int b  = m0 / HW;
  int p0 = m0 - b * HW;            // tile never crosses a batch (HW % 16 == 0)
  int n  = lane & 15;
  int mr = (lane & 16) ? 8 : 0;    // D layout: vgpr g holds row mr+g, col n
  float* base = Y + (size_t)(b * OUTC) * HW + p0 + mr;
  *(v8f*)(base + (size_t)(n)      * HW) = acc0;
  *(v8f*)(base + (size_t)(n + 16) * HW) = acc1;
  *(v8f*)(base + (size_t)(n + 32) * HW) = acc2;
  *(v8f*)(base + (size_t)(n + 48) * HW) = acc3;
}

// ---------------------------------------------------------------------------
// Bilinear align_corners=True upsample of y2/y3/y4 (already conv'ed, 64ch)
// added in place onto out (which holds W1@x1).
// ---------------------------------------------------------------------------
__device__ __forceinline__ float bilin_ac(const float* __restrict__ y, int b, int c,
                                          int s, int h, int w) {
  float f = (float)(s - 1) / 127.0f;
  float fy = h * f;
  float fx = w * f;
  int y0 = (int)fy; int y1 = (y0 + 1 < s) ? y0 + 1 : s - 1; float wy = fy - y0;
  int x0 = (int)fx; int x1 = (x0 + 1 < s) ? x0 + 1 : s - 1; float wx = fx - x0;
  const float* base = y + (size_t)(b * OUTC + c) * s * s;
  float v00 = base[y0 * s + x0], v01 = base[y0 * s + x1];
  float v10 = base[y1 * s + x0], v11 = base[y1 * s + x1];
  return (v00 * (1.f - wx) + v01 * wx) * (1.f - wy) + (v10 * (1.f - wx) + v11 * wx) * wy;
}

__global__ void upsample_add_kernel(float* __restrict__ out,
                                    const float* __restrict__ y2,
                                    const float* __restrict__ y3,
                                    const float* __restrict__ y4) {
  int idx = blockIdx.x * blockDim.x + threadIdx.x;
  if (idx >= NB * OUTC * HW0) return;
  int w = idx & 127;
  int h = (idx >> 7) & 127;
  int c = (idx >> 14) & 63;
  int b = idx >> 20;
  out[idx] += bilin_ac(y2, b, c, 64, h, w) + bilin_ac(y3, b, c, 32, h, w) +
              bilin_ac(y4, b, c, 16, h, w);
}

// ---------------------------------------------------------------------------
// Strip pooling: hs[b][w][c] = mean_h out[b,c,h,w]; vs[b][h][c] = max_w ...
// ---------------------------------------------------------------------------
__global__ void pool_kernel(const float* __restrict__ out,
                            float* __restrict__ hs, float* __restrict__ vs) {
  int idx = blockIdx.x * blockDim.x + threadIdx.x;
  if (idx >= NB * OUTC * 128) return;
  int t = idx & 127;
  int c = (idx >> 7) & 63;
  int b = idx >> 13;
  const float* base = out + ((size_t)(b * OUTC + c) << 14);
  if (blockIdx.y == 0) {
    float s = 0.f;
    for (int h = 0; h < 128; ++h) s += base[h * 128 + t];
    hs[((b * 128) + t) * OUTC + c] = s * (1.0f / 128.0f);
  } else {
    float m = -3.0e38f;
    for (int w = 0; w < 128; ++w) m = fmaxf(m, base[t * 128 + w]);
    vs[((b * 128) + t) * OUTC + c] = m;
  }
}

// ---------------------------------------------------------------------------
// QKV projection: Q = s @ w1^T + b1  (tokens (B,128,64), weights (64,64))
// ---------------------------------------------------------------------------
__global__ void qkv_kernel(const float* __restrict__ s,
                           const float* __restrict__ w1, const float* __restrict__ bb1,
                           const float* __restrict__ w2, const float* __restrict__ bb2,
                           const float* __restrict__ w3, const float* __restrict__ bb3,
                           float* __restrict__ Q, float* __restrict__ K,
                           float* __restrict__ V) {
  int idx = blockIdx.x * blockDim.x + threadIdx.x;
  if (idx >= 3 * NB * 128 * OUTC) return;
  int c = idx & 63;
  int l = (idx >> 6) & 127;
  int b = (idx >> 13) & 3;
  int which = idx >> 15;
  const float* w  = (which == 0) ? w1  : (which == 1) ? w2  : w3;
  const float* bi = (which == 0) ? bb1 : (which == 1) ? bb2 : bb3;
  float*       o  = (which == 0) ? Q   : (which == 1) ? K   : V;
  const float* srow = s + ((size_t)(b * 128) + l) * OUTC;
  const float* wrow = w + c * OUTC;
  float acc = bi[c];
  for (int k = 0; k < OUTC; ++k) acc += srow[k] * wrow[k];
  o[((b * 128) + l) * OUTC + c] = acc;
}

// ---------------------------------------------------------------------------
// softmax(Q K^T / 8) @ V  — one block per (batch, query token)
// ---------------------------------------------------------------------------
__global__ void attn_kernel(const float* __restrict__ Q, const float* __restrict__ K,
                            const float* __restrict__ V, float* __restrict__ O) {
  __shared__ float sc[128];
  __shared__ float red[128];
  int bl = blockIdx.x;           // 0..511
  int b  = bl >> 7;
  int m  = threadIdx.x;          // key index
  const float* q    = Q + (size_t)bl * OUTC;
  const float* krow = K + ((size_t)(b * 128) + m) * OUTC;
  float d = 0.f;
  for (int k = 0; k < OUTC; ++k) d += q[k] * krow[k];
  sc[m] = d * 0.125f;            // 1/sqrt(64)
  red[m] = sc[m];
  __syncthreads();
  for (int s = 64; s > 0; s >>= 1) {
    if (m < s) red[m] = fmaxf(red[m], red[m + s]);
    __syncthreads();
  }
  float mx = red[0];
  __syncthreads();
  float e = expf(sc[m] - mx);
  sc[m] = e;
  red[m] = e;
  __syncthreads();
  for (int s = 64; s > 0; s >>= 1) {
    if (m < s) red[m] += red[m + s];
    __syncthreads();
  }
  float inv = 1.0f / red[0];
  if (m < OUTC) {
    float o = 0.f;
    for (int t = 0; t < 128; ++t) o += sc[t] * V[((b * 128) + t) * OUTC + m];
    O[(size_t)bl * OUTC + m] = o * inv;
  }
}

// ---------------------------------------------------------------------------
// out += h_attn (broadcast over H) + v_attn (broadcast over W)
// ---------------------------------------------------------------------------
__global__ void add_attn_kernel(float* __restrict__ out,
                                const float* __restrict__ ha,
                                const float* __restrict__ va) {
  int idx = blockIdx.x * blockDim.x + threadIdx.x;
  if (idx >= NB * OUTC * HW0) return;
  int w = idx & 127;
  int h = (idx >> 7) & 127;
  int c = (idx >> 14) & 63;
  int b = idx >> 20;
  out[idx] += ha[((b * 128) + w) * OUTC + c] + va[((b * 128) + h) * OUTC + c];
}

// ---------------------------------------------------------------------------
// Antialiased bilinear downsample 128 -> O (triangle kernel scaled by O/128,
// per-dimension weight normalization == jax.image.resize(antialias=True))
// ---------------------------------------------------------------------------
__global__ void downsample_kernel(const float* __restrict__ in, float* __restrict__ dst,
                                  int O, int total) {
  int idx = blockIdx.x * blockDim.x + threadIdx.x;
  if (idx >= total) return;
  int x = idx % O;
  int t = idx / O;
  int y = t % O;
  int bc = t / O;                    // 0..255 (b*64+c)
  float inv = 128.0f / (float)O;     // kernel support radius in input units
  float scale = (float)O / 128.0f;
  float cy = (y + 0.5f) * inv - 0.5f;
  float cx = (x + 0.5f) * inv - 0.5f;
  int ylo = (int)ceilf(cy - inv); if (ylo < 0) ylo = 0;
  int yhi = (int)floorf(cy + inv); if (yhi > 127) yhi = 127;
  int xlo = (int)ceilf(cx - inv); if (xlo < 0) xlo = 0;
  int xhi = (int)floorf(cx + inv); if (xhi > 127) xhi = 127;
  const float* base = in + (size_t)bc * HW0;
  float sumx = 0.f;
  for (int jx = xlo; jx <= xhi; ++jx) sumx += fmaxf(0.f, 1.f - fabsf(jx - cx) * scale);
  float sumy = 0.f, acc = 0.f;
  for (int jy = ylo; jy <= yhi; ++jy) {
    float wy = fmaxf(0.f, 1.f - fabsf(jy - cy) * scale);
    sumy += wy;
    const float* rowp = base + jy * 128;
    float racc = 0.f;
    for (int jx = xlo; jx <= xhi; ++jx) {
      float wx = fmaxf(0.f, 1.f - fabsf(jx - cx) * scale);
      racc += wx * rowp[jx];
    }
    acc += wy * racc;
  }
  dst[idx] = acc / (sumy * sumx);
}

// ---------------------------------------------------------------------------
extern "C" void kernel_launch(void* const* d_in, const int* in_sizes, int n_in,
                              void* d_out, int out_size, void* d_ws, size_t ws_size,
                              hipStream_t stream) {
  const float* x1 = (const float*)d_in[0];
  const float* x2 = (const float*)d_in[1];
  const float* x3 = (const float*)d_in[2];
  const float* x4 = (const float*)d_in[3];
  const float* cw = (const float*)d_in[4];
  const float* hw1 = (const float*)d_in[5];  const float* hb1 = (const float*)d_in[6];
  const float* hw2 = (const float*)d_in[7];  const float* hb2 = (const float*)d_in[8];
  const float* hw3 = (const float*)d_in[9];  const float* hb3 = (const float*)d_in[10];
  const float* vw1 = (const float*)d_in[11]; const float* vb1 = (const float*)d_in[12];
  const float* vw2 = (const float*)d_in[13]; const float* vb2 = (const float*)d_in[14];
  const float* vw3 = (const float*)d_in[15]; const float* vb3 = (const float*)d_in[16];

  float* out = (float*)d_out;                       // (4,64,128,128) NCHW
  float* out2 = out + (size_t)NB * OUTC * HW0;      // (4,64,64,64)
  float* out3 = out2 + (size_t)NB * OUTC * 64 * 64; // (4,64,32,32)
  float* out4 = out3 + (size_t)NB * OUTC * 32 * 32; // (4,64,16,16)

  char* ws = (char*)d_ws;
  size_t off = 0;
  auto take = [&](size_t bytes) -> void* {
    void* p = (void*)(ws + off);
    off += (bytes + 255) & ~(size_t)255;
    return p;
  };
  _Float16* A1 = (_Float16*)take((size_t)65536 * 64 * 2);
  _Float16* A2 = (_Float16*)take((size_t)16384 * 128 * 2);
  _Float16* A3 = (_Float16*)take((size_t)4096 * 256 * 2);
  _Float16* A4 = (_Float16*)take((size_t)1024 * 512 * 2);
  _Float16* Wh = (_Float16*)take((size_t)64 * 960 * 2);
  float* y2 = (float*)take((size_t)NB * OUTC * 4096 * 4);
  float* y3 = (float*)take((size_t)NB * OUTC * 1024 * 4);
  float* y4 = (float*)take((size_t)NB * OUTC * 256 * 4);
  float* hs = (float*)take((size_t)NB * 128 * OUTC * 4);
  float* vs = (float*)take((size_t)NB * 128 * OUTC * 4);
  float* Qh = (float*)take((size_t)NB * 128 * OUTC * 4);
  float* Kh = (float*)take((size_t)NB * 128 * OUTC * 4);
  float* Vh = (float*)take((size_t)NB * 128 * OUTC * 4);
  float* Qv = (float*)take((size_t)NB * 128 * OUTC * 4);
  float* Kv = (float*)take((size_t)NB * 128 * OUTC * 4);
  float* Vv = (float*)take((size_t)NB * 128 * OUTC * 4);
  float* ha = (float*)take((size_t)NB * 128 * OUTC * 4);
  float* va = (float*)take((size_t)NB * 128 * OUTC * 4);

  const int TPB = 256;
  auto blocks = [](int total, int tpb) { return (total + tpb - 1) / tpb; };

  // 1) pack A matrices (NCHW f32 -> [M][K] f16, LDS-tiled transpose) + weights
  pack_a_kernel<<<dim3(NB * (16384 / 32), 64 / 64), TPB, 0, stream>>>(x1, A1, 64, 16384);
  pack_a_kernel<<<dim3(NB * (4096 / 32), 128 / 64), TPB, 0, stream>>>(x2, A2, 128, 4096);
  pack_a_kernel<<<dim3(NB * (1024 / 32), 256 / 64), TPB, 0, stream>>>(x3, A3, 256, 1024);
  pack_a_kernel<<<dim3(NB * (256 / 32), 512 / 64), TPB, 0, stream>>>(x4, A4, 512, 256);
  pack_w_kernel<<<blocks(64 * 960, TPB), TPB, 0, stream>>>(cw, Wh, 64 * 960);

  // 2) per-scale 1x1 convs via WMMA (conv and bilinear upsample commute)
  conv1x1_wmma_kernel<<<65536 / 128, 256, 0, stream>>>(A1, Wh, out, 64, 0, 16384);
  conv1x1_wmma_kernel<<<16384 / 128, 256, 0, stream>>>(A2, Wh, y2, 128, 64, 4096);
  conv1x1_wmma_kernel<<<4096 / 128, 256, 0, stream>>>(A3, Wh, y3, 256, 192, 1024);
  conv1x1_wmma_kernel<<<1024 / 128, 256, 0, stream>>>(A4, Wh, y4, 512, 448, 256);

  // 3) out += up(y2)+up(y3)+up(y4)  (align_corners=True)
  upsample_add_kernel<<<blocks(NB * OUTC * HW0, TPB), TPB, 0, stream>>>(out, y2, y3, y4);

  // 4) strip pooling (mean over H -> hs; max over W -> vs)
  pool_kernel<<<dim3(blocks(NB * OUTC * 128, TPB), 2), TPB, 0, stream>>>(out, hs, vs);

  // 5) strip attentions
  qkv_kernel<<<blocks(3 * NB * 128 * OUTC, TPB), TPB, 0, stream>>>(
      hs, hw1, hb1, hw2, hb2, hw3, hb3, Qh, Kh, Vh);
  qkv_kernel<<<blocks(3 * NB * 128 * OUTC, TPB), TPB, 0, stream>>>(
      vs, vw1, vb1, vw2, vb2, vw3, vb3, Qv, Kv, Vv);
  attn_kernel<<<NB * 128, 128, 0, stream>>>(Qh, Kh, Vh, ha);
  attn_kernel<<<NB * 128, 128, 0, stream>>>(Qv, Kv, Vv, va);

  // 6) out += broadcasts
  add_attn_kernel<<<blocks(NB * OUTC * HW0, TPB), TPB, 0, stream>>>(out, ha, va);

  // 7) antialiased downsamples
  downsample_kernel<<<blocks(NB * OUTC * 64 * 64, TPB), TPB, 0, stream>>>(
      out, out2, 64, NB * OUTC * 64 * 64);
  downsample_kernel<<<blocks(NB * OUTC * 32 * 32, TPB), TPB, 0, stream>>>(
      out, out3, 32, NB * OUTC * 32 * 32);
  downsample_kernel<<<blocks(NB * OUTC * 16 * 16, TPB), TPB, 0, stream>>>(
      out, out4, 16, NB * OUTC * 16 * 16);
}
